// SelfAttention_32993938768140
// MI455X (gfx1250) — matrix operand order repaired and checked by reading, not compiled, not run on hardware
//
#include <hip/hip_runtime.h>
#include <hip/hip_bf16.h>
#include <stdint.h>

typedef __attribute__((ext_vector_type(16))) __bf16 bf16x16;
typedef __attribute__((ext_vector_type(8)))  float  f32x8;

union FragBF { bf16x16 v; uint4 q[2]; };

#define NEGV (-1000000000.0f)

__device__ __forceinline__ unsigned short f2bf(float f) {
    unsigned u = __float_as_uint(f);
    u += 0x7FFFu + ((u >> 16) & 1u);          // round-to-nearest-even
    return (unsigned short)(u >> 16);
}

// Async copy of 64 contiguous bytes global -> LDS (per-lane), ASYNCcnt-tracked.
// INST_OFFSET applies to both the LDS and global address, so contiguous
// segments need only one LDS addr + one global addr.
__device__ __forceinline__ void async_copy64B(unsigned ldsaddr, const void* gptr) {
    asm volatile(
        "global_load_async_to_lds_b128 %0, %1, off\n\t"
        "global_load_async_to_lds_b128 %0, %1, off offset:16\n\t"
        "global_load_async_to_lds_b128 %0, %1, off offset:32\n\t"
        "global_load_async_to_lds_b128 %0, %1, off offset:48"
        :: "v"(ldsaddr), "v"(gptr)
        : "memory");
}

__device__ __forceinline__ void wait_async_all() {
    asm volatile("s_wait_asynccnt 0x0" ::: "memory");
}

// Generic (flat) shared pointer -> 32-bit LDS address (aperture: low 32 bits)
__device__ __forceinline__ unsigned lds_addr(const void* p) {
    return (unsigned)(size_t)p;
}

// ---------------------------------------------------------------------------
// Transpose + convert weights: W[E][128] f32  ->  WT[128][E] bf16 (N-major)
// ---------------------------------------------------------------------------
__global__ __launch_bounds__(256)
void transpose_w_kernel(const float* __restrict__ w, unsigned short* __restrict__ wt) {
    int j = blockIdx.x * 256 + threadIdx.x;   // j < 128*1024
    int n = j >> 10;                          // / E(=1024)
    int e = j & 1023;
    wt[j] = f2bf(w[e * 128 + n]);
}

// ---------------------------------------------------------------------------
// Fused QKV projection GEMM: X[S,E] @ W[E,128] (+bias, Q scaled by 1/sqrt(128))
// grid = (S/64, B, 3); block = 128 (4 waves). z: 0=Q, 1=K, 2=V(transposed out)
// ---------------------------------------------------------------------------
__global__ __launch_bounds__(128)
void qkv_proj_kernel(const float* __restrict__ emb, const unsigned short* __restrict__ WT,
                     const float* __restrict__ bq, const float* __restrict__ bk,
                     const float* __restrict__ bv,
                     unsigned short* __restrict__ Qb, unsigned short* __restrict__ Kb,
                     unsigned short* __restrict__ Vt) {
    const int S = 2048, E = 1024, Dd = 128;
    const int s0 = blockIdx.x * 64, b = blockIdx.y, z = blockIdx.z;
    const float* X = emb + (size_t)b * S * E;
    const unsigned short* W = WT + (size_t)z * Dd * E;
    const float* bias = (z == 0) ? bq : ((z == 1) ? bk : bv);

    __shared__ __align__(16) unsigned short Xs[64][40];    // 32 K + 8 pad
    __shared__ __align__(16) unsigned short Ws[128][40];

    const int tid = threadIdx.x, wave = tid >> 5, lane = tid & 31;
    const int hf = lane >> 4, l16 = lane & 15;

    f32x8 acc[8];
#pragma unroll
    for (int t = 0; t < 8; ++t)
#pragma unroll
        for (int r = 0; r < 8; ++r) acc[t][r] = 0.f;

    for (int e0 = 0; e0 < E; e0 += 32) {
        __syncthreads();
        // W tile [128][32] bf16 (N-major): async global->LDS, 64B per thread
        async_copy64B(lds_addr(&Ws[tid][0]), W + (size_t)tid * E + e0);
        {   // X tile [64][32]: convert f32 -> bf16 into LDS, 16 floats/thread
            int r = tid >> 1, c = (tid & 1) * 16;
            const float4* src = (const float4*)(X + (size_t)(s0 + r) * E + e0 + c);
            float4 x0 = src[0], x1 = src[1], x2 = src[2], x3 = src[3];
            uint4 d0, d1;
            d0.x = f2bf(x0.x) | ((unsigned)f2bf(x0.y) << 16);
            d0.y = f2bf(x0.z) | ((unsigned)f2bf(x0.w) << 16);
            d0.z = f2bf(x1.x) | ((unsigned)f2bf(x1.y) << 16);
            d0.w = f2bf(x1.z) | ((unsigned)f2bf(x1.w) << 16);
            d1.x = f2bf(x2.x) | ((unsigned)f2bf(x2.y) << 16);
            d1.y = f2bf(x2.z) | ((unsigned)f2bf(x2.w) << 16);
            d1.z = f2bf(x3.x) | ((unsigned)f2bf(x3.y) << 16);
            d1.w = f2bf(x3.z) | ((unsigned)f2bf(x3.w) << 16);
            *(uint4*)&Xs[r][c] = d0;
            *(uint4*)&Xs[r][c + 8] = d1;
        }
        wait_async_all();
        __syncthreads();

        FragBF af;   // A frag: rows 16*wave .. +15
        af.q[0] = *(const uint4*)&Xs[16 * wave + l16][8 * hf];
        af.q[1] = *(const uint4*)&Xs[16 * wave + l16][16 + 8 * hf];
#pragma unroll
        for (int t = 0; t < 8; ++t) {
            FragBF bfr;
            bfr.q[0] = *(const uint4*)&Ws[16 * t + l16][8 * hf];
            bfr.q[1] = *(const uint4*)&Ws[16 * t + l16][16 + 8 * hf];
            acc[t] = __builtin_amdgcn_wmma_f32_16x16x32_bf16(
                false, af.v, false, bfr.v, (short)0, acc[t], false, false);
        }
    }

    const float scale = (z == 0) ? 0.08838834764831845f : 1.0f;  // 1/sqrt(128)
    const int srow = s0 + 16 * wave;
    if (z < 2) {
        unsigned short* O = (z == 0 ? Qb : Kb) + ((size_t)b * S + srow) * Dd;
#pragma unroll
        for (int t = 0; t < 8; ++t) {
            float bn = bias[16 * t + l16];
#pragma unroll
            for (int r = 0; r < 8; ++r) {
                float v = (acc[t][r] + bn) * scale;
                O[(size_t)(8 * hf + r) * Dd + 16 * t + l16] = f2bf(v);
            }
        }
    } else {
        // Vt[b][d][s]: per lane, d fixed = 16t+l16, s contiguous -> one b128 store
#pragma unroll
        for (int t = 0; t < 8; ++t) {
            float bn = bias[16 * t + l16];
            uint4 pk;
            pk.x = f2bf(acc[t][0] + bn) | ((unsigned)f2bf(acc[t][1] + bn) << 16);
            pk.y = f2bf(acc[t][2] + bn) | ((unsigned)f2bf(acc[t][3] + bn) << 16);
            pk.z = f2bf(acc[t][4] + bn) | ((unsigned)f2bf(acc[t][5] + bn) << 16);
            pk.w = f2bf(acc[t][6] + bn) | ((unsigned)f2bf(acc[t][7] + bn) << 16);
            *(uint4*)(Vt + ((size_t)b * Dd + 16 * t + l16) * S + srow + 8 * hf) = pk;
        }
    }
}

// ---------------------------------------------------------------------------
// Flash attention: grid = (S/64, B); block = 128 (4 waves x 16 query rows)
// ---------------------------------------------------------------------------
__global__ __launch_bounds__(128)
void attn_kernel(const unsigned short* __restrict__ Qb, const unsigned short* __restrict__ Kb,
                 const unsigned short* __restrict__ Vt, const int* __restrict__ mask,
                 float* __restrict__ out) {
    const int S = 2048, Dd = 128;
    const int q0 = blockIdx.x * 64, b = blockIdx.y;
    const unsigned short* Q = Qb + (size_t)b * S * Dd;
    const unsigned short* K = Kb + (size_t)b * S * Dd;
    const unsigned short* V = Vt + (size_t)b * Dd * S;
    const int* mk = mask + (size_t)b * S;

    __shared__ __align__(16) unsigned short Ks[32][136];   // [kv][d], 8-elt pad
    __shared__ __align__(16) unsigned short Vs[128][40];   // [d][kv] (N-major for PV)
    __shared__ __align__(16) unsigned short Ps[4][16][40]; // per-wave P bounce

    const int tid = threadIdx.x, wave = tid >> 5, lane = tid & 31;
    const int hf = lane >> 4, l16 = lane & 15;
    const int qbase = q0 + 16 * wave;

    FragBF qf[4];   // Q row frags for 4 K-chunks of 32
    {
        const unsigned short* qrow = Q + (size_t)(qbase + l16) * Dd;
#pragma unroll
        for (int c = 0; c < 4; ++c) {
            qf[c].q[0] = *(const uint4*)(qrow + 32 * c + 8 * hf);
            qf[c].q[1] = *(const uint4*)(qrow + 32 * c + 16 + 8 * hf);
        }
    }

    float mrow[8], lrow[8];
    f32x8 oacc[8];
#pragma unroll
    for (int r = 0; r < 8; ++r) { mrow[r] = -3.0e38f; lrow[r] = 0.f; }
#pragma unroll
    for (int t = 0; t < 8; ++t)
#pragma unroll
        for (int r = 0; r < 8; ++r) oacc[t][r] = 0.f;

    const int kv_end = q0 + 64;   // causal bound for this workgroup
    for (int kv0 = 0; kv0 < kv_end; kv0 += 32) {
        __syncthreads();
        {   // K tile [32][128] + V^T tile [128][32]: async global->LDS
            int r = tid >> 2, c = (tid & 3) * 32;
            async_copy64B(lds_addr(&Ks[r][c]), K + (size_t)(kv0 + r) * Dd + c);
            async_copy64B(lds_addr(&Vs[tid][0]), V + (size_t)tid * S + kv0);
        }
        wait_async_all();
        __syncthreads();

        f32x8 sc0, sc1;
#pragma unroll
        for (int r = 0; r < 8; ++r) { sc0[r] = 0.f; sc1[r] = 0.f; }
#pragma unroll
        for (int c = 0; c < 4; ++c) {
            FragBF k0, k1;
            k0.q[0] = *(const uint4*)&Ks[l16][32 * c + 8 * hf];
            k0.q[1] = *(const uint4*)&Ks[l16][32 * c + 16 + 8 * hf];
            k1.q[0] = *(const uint4*)&Ks[16 + l16][32 * c + 8 * hf];
            k1.q[1] = *(const uint4*)&Ks[16 + l16][32 * c + 16 + 8 * hf];
            sc0 = __builtin_amdgcn_wmma_f32_16x16x32_bf16(
                false, qf[c].v, false, k0.v, (short)0, sc0, false, false);
            sc1 = __builtin_amdgcn_wmma_f32_16x16x32_bf16(
                false, qf[c].v, false, k1.v, (short)0, sc1, false, false);
        }

        const int kvA = kv0 + l16, kvB = kv0 + 16 + l16;
        const bool vA = (mk[kvA] != 0), vB = (mk[kvB] != 0);

        float pa[8], pb[8];
#pragma unroll
        for (int r = 0; r < 8; ++r) {
            int qr = qbase + 8 * hf + r;
            float a  = ((kvA <= qr) && vA) ? sc0[r] : NEGV;
            float bb = ((kvB <= qr) && vB) ? sc1[r] : NEGV;
            float bm = fmaxf(a, bb);
#pragma unroll
            for (int off = 1; off < 16; off <<= 1)
                bm = fmaxf(bm, __shfl_xor(bm, off, 32));     // stays in 16-lane half
            float mnew  = fmaxf(mrow[r], bm);
            float alpha = __expf(mrow[r] - mnew);
            float ea = __expf(a - mnew);
            float eb = __expf(bb - mnew);
            float bs = ea + eb;
#pragma unroll
            for (int off = 1; off < 16; off <<= 1)
                bs += __shfl_xor(bs, off, 32);
            lrow[r] = lrow[r] * alpha + bs;
            mrow[r] = mnew;
            pa[r] = ea; pb[r] = eb;
#pragma unroll
            for (int t = 0; t < 8; ++t) oacc[t][r] *= alpha;
        }

        // D-layout -> A-layout via wave-private LDS bounce (bf16)
#pragma unroll
        for (int r = 0; r < 8; ++r) {
            Ps[wave][8 * hf + r][l16]      = f2bf(pa[r]);
            Ps[wave][8 * hf + r][16 + l16] = f2bf(pb[r]);
        }
        asm volatile("s_wait_dscnt 0" ::: "memory");   // same-wave LDS RAW

        FragBF pf;
        pf.q[0] = *(const uint4*)&Ps[wave][l16][8 * hf];
        pf.q[1] = *(const uint4*)&Ps[wave][l16][16 + 8 * hf];
#pragma unroll
        for (int t = 0; t < 8; ++t) {
            FragBF vf;
            vf.q[0] = *(const uint4*)&Vs[16 * t + l16][8 * hf];
            vf.q[1] = *(const uint4*)&Vs[16 * t + l16][16 + 8 * hf];
            oacc[t] = __builtin_amdgcn_wmma_f32_16x16x32_bf16(
                false, pf.v, false, vf.v, (short)0, oacc[t], false, false);
        }
    }

    float* orow = out + ((size_t)b * S + qbase) * Dd;
#pragma unroll
    for (int r = 0; r < 8; ++r) {
        float inv = 1.0f / lrow[r];
#pragma unroll
        for (int t = 0; t < 8; ++t)
            orow[(size_t)(8 * hf + r) * Dd + 16 * t + l16] = oacc[t][r] * inv;
    }
}

// ---------------------------------------------------------------------------
extern "C" void kernel_launch(void* const* d_in, const int* in_sizes, int n_in,
                              void* d_out, int out_size, void* d_ws, size_t ws_size,
                              hipStream_t stream) {
    (void)in_sizes; (void)n_in; (void)out_size; (void)ws_size;
    const float* emb  = (const float*)d_in[0];
    const int*   mask = (const int*)d_in[1];
    const float* Wq   = (const float*)d_in[2];
    const float* bq   = (const float*)d_in[3];
    const float* Wk   = (const float*)d_in[4];
    const float* bk   = (const float*)d_in[5];
    const float* Wv   = (const float*)d_in[6];
    const float* bv   = (const float*)d_in[7];
    float* out = (float*)d_out;

    const int B = 8, S = 2048, E = 1024, Dd = 128;
    char* ws = (char*)d_ws;
    unsigned short* WT = (unsigned short*)ws;                        // 3*128*1024 bf16
    unsigned short* Qb = (unsigned short*)(ws + (size_t)3 * Dd * E * 2);
    unsigned short* Kb = Qb + (size_t)B * S * Dd;
    unsigned short* Vt = Kb + (size_t)B * S * Dd;                    // total ~12.8 MB

    transpose_w_kernel<<<512, 256, 0, stream>>>(Wq, WT);
    transpose_w_kernel<<<512, 256, 0, stream>>>(Wk, WT + Dd * E);
    transpose_w_kernel<<<512, 256, 0, stream>>>(Wv, WT + 2 * Dd * E);
    qkv_proj_kernel<<<dim3(S / 64, B, 3), 128, 0, stream>>>(emb, WT, bq, bk, bv, Qb, Kb, Vt);
    attn_kernel<<<dim3(S / 64, B), 128, 0, stream>>>(Qb, Kb, Vt, mask, out);
}
